// AlignHead_85220741088089
// MI455X (gfx1250) — matrix operand order repaired and checked by reading, not compiled, not run on hardware
//
#include <hip/hip_runtime.h>

typedef __bf16 bf16_t;
typedef __attribute__((ext_vector_type(16))) __bf16 v16bf;
typedef __attribute__((ext_vector_type(8)))  float  v8f;
typedef __attribute__((address_space(3))) bf16_t lds_bf16;

#define N_ENT   50000
#define N_EDGE  500000
#define D_HID   256
#define H1      8
#define C1      64
#define F1      512          // H1*C1
#define NEG_SLOPE 0.2f
#define LN_EPS  1e-5f
#define E_TOT   (N_EDGE + N_ENT)   // edges + self loops

// ---------- helpers ----------
__device__ __forceinline__ unsigned fenc(float f) {
  unsigned u = __float_as_uint(f);
  return (u & 0x80000000u) ? ~u : (u | 0x80000000u);
}
__device__ __forceinline__ float fdec(unsigned u) {
  return __uint_as_float((u & 0x80000000u) ? (u & 0x7fffffffu) : ~u);
}

// ---------- 1) LayerNorm -> bf16 ----------
__global__ void k_layernorm(const float* __restrict__ x,
                            const float* __restrict__ w,
                            const float* __restrict__ b,
                            bf16_t* __restrict__ out) {
  int wid  = (blockIdx.x * blockDim.x + threadIdx.x) >> 5;
  int lane = threadIdx.x & 31;
  if (wid >= N_ENT) return;
  const float4* row = (const float4*)(x + (size_t)wid * D_HID);
  float4 v0 = row[lane];
  float4 v1 = row[lane + 32];
  float s = v0.x + v0.y + v0.z + v0.w + v1.x + v1.y + v1.z + v1.w;
  float q = v0.x*v0.x + v0.y*v0.y + v0.z*v0.z + v0.w*v0.w
          + v1.x*v1.x + v1.y*v1.y + v1.z*v1.z + v1.w*v1.w;
  #pragma unroll
  for (int m = 16; m; m >>= 1) { s += __shfl_xor(s, m, 32); q += __shfl_xor(q, m, 32); }
  float mu  = s * (1.0f / D_HID);
  float var = q * (1.0f / D_HID) - mu * mu;
  float r   = rsqrtf(var + LN_EPS);
  bf16_t* orow = out + (size_t)wid * D_HID;
  int i0 = lane * 4, i1 = (lane + 32) * 4;
  orow[i0+0] = (bf16_t)((v0.x - mu) * r * w[i0+0] + b[i0+0]);
  orow[i0+1] = (bf16_t)((v0.y - mu) * r * w[i0+1] + b[i0+1]);
  orow[i0+2] = (bf16_t)((v0.z - mu) * r * w[i0+2] + b[i0+2]);
  orow[i0+3] = (bf16_t)((v0.w - mu) * r * w[i0+3] + b[i0+3]);
  orow[i1+0] = (bf16_t)((v1.x - mu) * r * w[i1+0] + b[i1+0]);
  orow[i1+1] = (bf16_t)((v1.y - mu) * r * w[i1+1] + b[i1+1]);
  orow[i1+2] = (bf16_t)((v1.z - mu) * r * w[i1+2] + b[i1+2]);
  orow[i1+3] = (bf16_t)((v1.w - mu) * r * w[i1+3] + b[i1+3]);
}

// ---------- 2) transpose + convert weights to bf16 : W[K][N] -> Wt[N][K] ----------
__global__ void k_convert_t(const float* __restrict__ W, bf16_t* __restrict__ Wt,
                            int K, int Nout) {
  int idx = blockIdx.x * blockDim.x + threadIdx.x;
  if (idx >= K * Nout) return;
  int n = idx / K, k = idx - n * K;
  Wt[(size_t)n * K + k] = (bf16_t)W[(size_t)k * Nout + n];
}

// ---------- 3) WMMA GEMM with async-LDS-staged B panel ----------
// block = 256 thr (8 waves). Block tile: 128(M) x 64(N). B panel (64 cols x K bf16)
// is DMA'd Global->LDS once per block via global_load_async_to_lds_b128 and
// shared by all 8 waves; each wave computes a 16x64 strip with 4 WMMAs/K-step.
template <int K>
__global__ __launch_bounds__(256) void k_gemm_bf16(const bf16_t* __restrict__ A,
                                                   const bf16_t* __restrict__ Bt,
                                                   float* __restrict__ C,
                                                   int M, int Nout) {
  __shared__ bf16_t sB[64 * K];            // 32KB (K=256) / 64KB (K=512)
  const int lane = threadIdx.x & 31;
  const int wv   = threadIdx.x >> 5;       // wave in block: 0..7
  const int mt   = blockIdx.x * 8 + wv;    // 16-row tile index
  const int ng   = blockIdx.y;             // 64-col group
  const int Mt   = (M + 15) >> 4;

  // ---- stage B panel: contiguous 64*K bf16 from Bt[ng*64 .. ng*64+63][*] ----
  {
    const char* gsrc = (const char*)(Bt + (size_t)ng * 64 * K);
    char* lbase = (char*)sB;
    constexpr int BYTES = 64 * K * 2;
    for (int c = threadIdx.x * 16; c < BYTES; c += 256 * 16) {
      lds_bf16* lp = (lds_bf16*)(lbase + c);
      unsigned long long ga = (unsigned long long)(gsrc + c);
      asm volatile("global_load_async_to_lds_b128 %0, %1, off"
                   :: "v"(lp), "v"(ga) : "memory");
    }
    asm volatile("s_wait_asynccnt 0" ::: "memory");
    __syncthreads();
  }

  if (mt < Mt) {
    const int lm = lane & 15, half = lane >> 4;
    int row = mt * 16 + lm; if (row >= M) row = M - 1;
    const bf16_t* arow = A + (size_t)row * K;
    const bf16_t* sb0 = sB + (size_t)(lm +  0) * K;
    const bf16_t* sb1 = sB + (size_t)(lm + 16) * K;
    const bf16_t* sb2 = sB + (size_t)(lm + 32) * K;
    const bf16_t* sb3 = sB + (size_t)(lm + 48) * K;

    v8f acc0 = {}, acc1 = {}, acc2 = {}, acc3 = {};
    #pragma unroll 4
    for (int k0 = 0; k0 < K; k0 += 32) {
      // A 16x32 bf16 layout: lane half h, element j -> K = j + 16*h
      v16bf a;
      #pragma unroll
      for (int j = 0; j < 16; ++j) {
        int kk = k0 + j + (half << 4);
        a[j] = arow[kk];
      }
      // B 32x16 bf16 layout: lane half h holds contiguous K = k0 + 16*h .. +15
      int kb = k0 + (half << 4);
      v16bf b0 = *(const v16bf*)(sb0 + kb);
      v16bf b1 = *(const v16bf*)(sb1 + kb);
      v16bf b2 = *(const v16bf*)(sb2 + kb);
      v16bf b3 = *(const v16bf*)(sb3 + kb);
      acc0 = __builtin_amdgcn_wmma_f32_16x16x32_bf16(false, a, false, b0, (short)0, acc0, false, false);
      acc1 = __builtin_amdgcn_wmma_f32_16x16x32_bf16(false, a, false, b1, (short)0, acc1, false, false);
      acc2 = __builtin_amdgcn_wmma_f32_16x16x32_bf16(false, a, false, b2, (short)0, acc2, false, false);
      acc3 = __builtin_amdgcn_wmma_f32_16x16x32_bf16(false, a, false, b3, (short)0, acc3, false, false);
    }
    // C/D layout: lane -> col = lane&15 ; VGPR v -> row = v + 8*(lane>>4)
    #pragma unroll
    for (int v = 0; v < 8; ++v) {
      int r = mt * 16 + v + (half << 3);
      if (r < M) {
        float* crow = C + (size_t)r * Nout + ng * 64 + lm;
        crow[ 0] = acc0[v];
        crow[16] = acc1[v];
        crow[32] = acc2[v];
        crow[48] = acc3[v];
      }
    }
  }
}

// ---------- 4) per-(node,head) attention dot products (conv1) ----------
__global__ void k_att1(const float* __restrict__ h1,
                       const float* __restrict__ att_src,
                       const float* __restrict__ att_dst,
                       float* __restrict__ as, float* __restrict__ ad) {
  int idx = blockIdx.x * blockDim.x + threadIdx.x;
  if (idx >= N_ENT * H1) return;
  int n = idx >> 3, h = idx & 7;
  const float4* hp = (const float4*)(h1 + (size_t)n * F1 + h * C1);
  const float4* sp = (const float4*)(att_src + h * C1);
  const float4* dp = (const float4*)(att_dst + h * C1);
  float ss = 0.f, sd = 0.f;
  #pragma unroll
  for (int t = 0; t < C1 / 4; ++t) {
    float4 hv = hp[t], sv = sp[t], dv = dp[t];
    ss += hv.x*sv.x + hv.y*sv.y + hv.z*sv.z + hv.w*sv.w;
    sd += hv.x*dv.x + hv.y*dv.y + hv.z*dv.z + hv.w*dv.w;
  }
  as[idx] = ss; ad[idx] = sd;
}

// ---------- 5) per-node attention (conv2, heads=1, D=256) ----------
__global__ void k_att2(const float* __restrict__ h2,
                       const float* __restrict__ att_src,
                       const float* __restrict__ att_dst,
                       float* __restrict__ as, float* __restrict__ ad) {
  int wid  = (blockIdx.x * blockDim.x + threadIdx.x) >> 5;
  int lane = threadIdx.x & 31;
  if (wid >= N_ENT) return;
  const float4* hp = (const float4*)(h2 + (size_t)wid * D_HID);
  const float4* sp = (const float4*)att_src;
  const float4* dp = (const float4*)att_dst;
  float ss = 0.f, sd = 0.f;
  #pragma unroll
  for (int t = 0; t < 2; ++t) {
    int i = t * 32 + lane;
    float4 hv = hp[i], sv = sp[i], dv = dp[i];
    ss += hv.x*sv.x + hv.y*sv.y + hv.z*sv.z + hv.w*sv.w;
    sd += hv.x*dv.x + hv.y*dv.y + hv.z*dv.z + hv.w*dv.w;
  }
  #pragma unroll
  for (int m = 16; m; m >>= 1) { ss += __shfl_xor(ss, m, 32); sd += __shfl_xor(sd, m, 32); }
  if (lane == 0) { as[wid] = ss; ad[wid] = sd; }
}

// ---------- 6) segment max / segment sum over edges ----------
__global__ void k_edge_max(const int* __restrict__ ei,
                           const float* __restrict__ as, const float* __restrict__ ad,
                           unsigned* __restrict__ m, int H) {
  int idx = blockIdx.x * blockDim.x + threadIdx.x;
  if (idx >= E_TOT * H) return;
  int e = idx / H, h = idx - e * H;
  int s, d;
  if (e < N_EDGE) { s = ei[e]; d = ei[N_EDGE + e]; } else { s = d = e - N_EDGE; }
  float v = as[s * H + h] + ad[d * H + h];
  v = v > 0.f ? v : NEG_SLOPE * v;
  atomicMax(&m[d * H + h], fenc(v));
}

__global__ void k_edge_sum(const int* __restrict__ ei,
                           const float* __restrict__ as, const float* __restrict__ ad,
                           const unsigned* __restrict__ m, float* __restrict__ den, int H) {
  int idx = blockIdx.x * blockDim.x + threadIdx.x;
  if (idx >= E_TOT * H) return;
  int e = idx / H, h = idx - e * H;
  int s, d;
  if (e < N_EDGE) { s = ei[e]; d = ei[N_EDGE + e]; } else { s = d = e - N_EDGE; }
  float v = as[s * H + h] + ad[d * H + h];
  v = v > 0.f ? v : NEG_SLOPE * v;
  atomicAdd(&den[d * H + h], __expf(v - fdec(m[d * H + h])));
}

// ---------- 7) weighted scatter aggregation: wave per edge ----------
template <int F, int HEADS>
__global__ void k_agg(const int* __restrict__ ei, const float* __restrict__ h,
                      const float* __restrict__ as, const float* __restrict__ ad,
                      const unsigned* __restrict__ m, const float* __restrict__ den,
                      float* __restrict__ out) {
  int wid  = (blockIdx.x * blockDim.x + threadIdx.x) >> 5;
  int lane = threadIdx.x & 31;
  if (wid >= E_TOT) return;
  int s, d;
  if (wid < N_EDGE) { s = ei[wid]; d = ei[N_EDGE + wid]; } else { s = d = wid - N_EDGE; }
  float alpha = 0.f;
  if (lane < HEADS) {
    float v = as[s * HEADS + lane] + ad[d * HEADS + lane];
    v = v > 0.f ? v : NEG_SLOPE * v;
    alpha = __expf(v - fdec(m[d * HEADS + lane])) / den[d * HEADS + lane];
  }
  const float4* hs = (const float4*)(h + (size_t)s * F);
  float* od = out + (size_t)d * F;
  #pragma unroll
  for (int it = 0; it < F / 128; ++it) {
    int f4 = it * 32 + lane;
    int f  = f4 * 4;
    int head = (HEADS == 1) ? 0 : (f >> 6);   // C1 = 64
    float al = __shfl(alpha, head, 32);
    float4 hv = hs[f4];
    atomicAdd(od + f + 0, hv.x * al);
    atomicAdd(od + f + 1, hv.y * al);
    atomicAdd(od + f + 2, hv.z * al);
    atomicAdd(od + f + 3, hv.w * al);
  }
}

// ---------- 8) bias + ELU -> bf16 ----------
__global__ void k_elu_bf16(const float* __restrict__ in, const float* __restrict__ bias,
                           bf16_t* __restrict__ out) {
  int idx = blockIdx.x * blockDim.x + threadIdx.x;
  if (idx >= N_ENT * F1) return;
  float t = in[idx] + bias[idx & (F1 - 1)];
  t = t > 0.f ? t : expm1f(t);
  out[idx] = (bf16_t)t;
}

// ---------- 9) final bias ----------
__global__ void k_bias(float* __restrict__ out, const float* __restrict__ b) {
  int idx = blockIdx.x * blockDim.x + threadIdx.x;
  if (idx >= N_ENT * D_HID) return;
  out[idx] += b[idx & (D_HID - 1)];
}

static inline size_t alignup(size_t x) { return (x + 255) & ~(size_t)255; }

extern "C" void kernel_launch(void* const* d_in, const int* in_sizes, int n_in,
                              void* d_out, int out_size, void* d_ws, size_t ws_size,
                              hipStream_t stream) {
  const int*   edge = (const int*)  d_in[0];
  const float* x    = (const float*)d_in[1];
  const float* lnw  = (const float*)d_in[2];
  const float* lnb  = (const float*)d_in[3];
  const float* W1   = (const float*)d_in[4];
  const float* as1w = (const float*)d_in[5];
  const float* ad1w = (const float*)d_in[6];
  const float* b1   = (const float*)d_in[7];
  const float* W2   = (const float*)d_in[8];
  const float* as2w = (const float*)d_in[9];
  const float* ad2w = (const float*)d_in[10];
  const float* b2   = (const float*)d_in[11];
  float* out = (float*)d_out;

  char* p = (char*)d_ws;
  auto grab = [&](size_t bytes) -> char* { char* r = p; p += alignup(bytes); return r; };
  bf16_t*   xn   = (bf16_t*)  grab((size_t)N_ENT * D_HID * 2);
  bf16_t*   w1t  = (bf16_t*)  grab((size_t)F1 * D_HID * 2);
  bf16_t*   w2t  = (bf16_t*)  grab((size_t)D_HID * F1 * 2);
  float*    h1   = (float*)   grab((size_t)N_ENT * F1 * 4);
  float*    sa1  = (float*)   grab((size_t)N_ENT * H1 * 4);
  float*    da1  = (float*)   grab((size_t)N_ENT * H1 * 4);
  unsigned* m1   = (unsigned*)grab((size_t)N_ENT * H1 * 4);
  float*    den1 = (float*)   grab((size_t)N_ENT * H1 * 4);
  float*    out1 = (float*)   grab((size_t)N_ENT * F1 * 4);
  bf16_t*   e1   = (bf16_t*)  grab((size_t)N_ENT * F1 * 2);
  float*    h2   = (float*)   grab((size_t)N_ENT * D_HID * 4);
  float*    sa2  = (float*)   grab((size_t)N_ENT * 4);
  float*    da2  = (float*)   grab((size_t)N_ENT * 4);
  unsigned* m2   = (unsigned*)grab((size_t)N_ENT * 4);
  float*    den2 = (float*)   grab((size_t)N_ENT * 4);

  // zero the accumulators (harness poisons ws/out, never re-zeros)
  hipMemsetAsync(m1,   0, (size_t)N_ENT * H1 * 4, stream);
  hipMemsetAsync(den1, 0, (size_t)N_ENT * H1 * 4, stream);
  hipMemsetAsync(out1, 0, (size_t)N_ENT * F1 * 4, stream);
  hipMemsetAsync(m2,   0, (size_t)N_ENT * 4, stream);
  hipMemsetAsync(den2, 0, (size_t)N_ENT * 4, stream);
  hipMemsetAsync(out,  0, (size_t)N_ENT * D_HID * 4, stream);

  const int B = 256;
  const int Mt      = (N_ENT + 15) / 16;   // 3125
  const int MBLOCKS = (Mt + 7) / 8;        // 391

  // weights -> bf16 transposed
  k_convert_t<<<(D_HID * F1 + B - 1) / B, B, 0, stream>>>(W1, w1t, D_HID, F1);
  k_convert_t<<<(F1 * D_HID + B - 1) / B, B, 0, stream>>>(W2, w2t, F1, D_HID);

  // layernorm
  k_layernorm<<<(N_ENT + 7) / 8, B, 0, stream>>>(x, lnw, lnb, xn);

  // conv1 GEMM: h1 = xn @ W1   (M=50000, K=256, N=512)
  k_gemm_bf16<D_HID><<<dim3(MBLOCKS, F1 / 64), B, 0, stream>>>(xn, w1t, h1, N_ENT, F1);

  // conv1 attention
  k_att1<<<(N_ENT * H1 + B - 1) / B, B, 0, stream>>>(h1, as1w, ad1w, sa1, da1);
  k_edge_max<<<((size_t)E_TOT * H1 + B - 1) / B, B, 0, stream>>>(edge, sa1, da1, m1, H1);
  k_edge_sum<<<((size_t)E_TOT * H1 + B - 1) / B, B, 0, stream>>>(edge, sa1, da1, m1, den1, H1);
  k_agg<F1, H1><<<(E_TOT + 7) / 8, B, 0, stream>>>(edge, h1, sa1, da1, m1, den1, out1);

  // bias + ELU -> bf16
  k_elu_bf16<<<((size_t)N_ENT * F1 + B - 1) / B, B, 0, stream>>>(out1, b1, e1);

  // conv2 GEMM: h2 = e1 @ W2   (M=50000, K=512, N=256)
  k_gemm_bf16<F1><<<dim3(MBLOCKS, D_HID / 64), B, 0, stream>>>(e1, w2t, h2, N_ENT, D_HID);

  // conv2 attention (1 head)
  k_att2<<<(N_ENT + 7) / 8, B, 0, stream>>>(h2, as2w, ad2w, sa2, da2);
  k_edge_max<<<(E_TOT + B - 1) / B, B, 0, stream>>>(edge, sa2, da2, m2, 1);
  k_edge_sum<<<(E_TOT + B - 1) / B, B, 0, stream>>>(edge, sa2, da2, m2, den2, 1);
  k_agg<D_HID, 1><<<(E_TOT + 7) / 8, B, 0, stream>>>(edge, h2, sa2, da2, m2, den2, out);

  // final bias
  k_bias<<<((size_t)N_ENT * D_HID + B - 1) / B, B, 0, stream>>>(out, b2);
}
